// CHUNKER_1477468750180
// MI455X (gfx1250) — matrix operand order repaired and checked by reading, not compiled, not run on hardware
//
#include <hip/hip_runtime.h>
#include <hip/hip_bf16.h>

typedef __attribute__((ext_vector_type(16))) _Float16 v16h;
typedef __attribute__((ext_vector_type(8)))  _Float16 v8h;
typedef __attribute__((ext_vector_type(8)))  float    v8f;

namespace {
constexpr int kNTok   = 384;
constexpr int kWdim   = 512;
constexpr int kEdim   = 1024;          // 2*WDIM
constexpr int kHdim   = 1024;
constexpr int kK3real = 1072;          // HDIM + 48 feats
constexpr int kK3     = 1088;          // padded to multiple of 32
constexpr int kSpan   = kNTok * (kNTok + 1) / 2;  // 73920
constexpr int kMT     = 32;            // spans per workgroup
constexpr int kStride = 1088;          // LDS row stride in halves

// workspace layout (bytes); f16 elements are 2 bytes
constexpr size_t kPrefOff   = 0;
constexpr size_t kPrefBytes = (size_t)(kNTok + 1) * kEdim * 4;
constexpr size_t kWt1Off    = kPrefOff + kPrefBytes;
constexpr size_t kWt1Bytes  = (size_t)kHdim * kEdim * 2;
constexpr size_t kWt2Off    = kWt1Off + kWt1Bytes;
constexpr size_t kWt2Bytes  = (size_t)kHdim * kHdim * 2;
constexpr size_t kWts1Off   = kWt2Off + kWt2Bytes;
}

// ---------------- kernel 1: embedding gather + column-wise prefix sums ----
__global__ void k_embed_prefix(const int* __restrict__ sentence,
                               const int* __restrict__ pos_tags,
                               const float* __restrict__ We_wrd,
                               const float* __restrict__ We_pos,
                               float* __restrict__ pref) {
  int c = blockIdx.x * blockDim.x + threadIdx.x;
  if (c >= kEdim) return;
  float s = 0.f;
  pref[c] = 0.f;
  for (int t = 0; t < kNTok; ++t) {
    float v = (c < kWdim) ? We_pos[(size_t)pos_tags[t] * kWdim + c]
                          : We_wrd[(size_t)sentence[t] * kWdim + (c - kWdim)];
    s += v;
    pref[(size_t)(t + 1) * kEdim + c] = s;
  }
}

// ---------------- kernel 2: transpose + f16-convert the GEMM weights ------
__global__ void k_prep_weights(const float* __restrict__ W1,
                               const float* __restrict__ W2,
                               const float* __restrict__ Ws1,
                               _Float16* __restrict__ Wt1,
                               _Float16* __restrict__ Wt2,
                               _Float16* __restrict__ Wts1) {
  const int step = gridDim.x * blockDim.x;
  const int t0 = blockIdx.x * blockDim.x + threadIdx.x;
  for (int idx = t0; idx < kHdim * kEdim; idx += step) {
    int o = idx >> 10, k = idx & 1023;
    Wt1[(size_t)o * kEdim + k] = (_Float16)W1[(size_t)k * kHdim + o];
  }
  for (int idx = t0; idx < kHdim * kHdim; idx += step) {
    int o = idx >> 10, k = idx & 1023;
    Wt2[(size_t)o * kHdim + k] = (_Float16)W2[(size_t)k * kHdim + o];
  }
  for (int idx = t0; idx < kHdim * kK3; idx += step) {
    int o = idx / kK3, k = idx % kK3;
    float v = (k < kK3real) ? Ws1[(size_t)k * kHdim + o] : 0.f;
    Wts1[(size_t)o * kK3 + k] = (_Float16)v;
  }
}

// ---- one fused WMMA GEMM stage: O[32,1024] = relu(A[32,ksz] @ Bt^T + bias)
// A, O in LDS (row stride kStride halves); Bt is [1024][ksz] f16 row-major.
// Register-blocked: 2 M-tiles x 4 N-tiles of accumulators per wave, K loop
// kept rolled so fragment live ranges stay short (no spills).
__device__ __forceinline__ void gemm_tile(const _Float16* __restrict__ A,
                                          const _Float16* __restrict__ Bt,
                                          const float* __restrict__ bias,
                                          _Float16* __restrict__ O,
                                          int ksz, int lane, int wave) {
  const int rlo   = lane & 15;
  const int offA  = (lane < 16) ? 0 : 8;    // A frag: K 0-7/16-23 vs 8-15/24-31
  const int offB  = (lane < 16) ? 0 : 8;    // B frag: K 0-7/16-23 vs 8-15/24-31
  const int mbase = (lane < 16) ? 0 : 8;    // D frag M base
  const _Float16* arow0 = A + rlo * kStride + offA;
  const _Float16* arow1 = arow0 + 16 * kStride;

  for (int nt0 = 0; nt0 < 8; nt0 += 4) {
    int ncol[4];
    const _Float16* bp[4];
    v8f acc0[4] = {};
    v8f acc1[4] = {};
#pragma unroll
    for (int j = 0; j < 4; ++j) {
      ncol[j] = (wave + (nt0 + j) * 8) * 16 + rlo;
      bp[j] = Bt + (size_t)ncol[j] * (size_t)ksz + offB;
    }
#pragma unroll 1
    for (int kb = 0; kb < ksz; kb += 32) {
      union U { v16h v; v8h h[2]; };
      U a0, a1;
      a0.h[0] = *(const v8h*)(arow0 + kb);
      a0.h[1] = *(const v8h*)(arow0 + kb + 16);
      a1.h[0] = *(const v8h*)(arow1 + kb);
      a1.h[1] = *(const v8h*)(arow1 + kb + 16);
#pragma unroll
      for (int j = 0; j < 4; ++j) {
        U b;
        b.h[0] = *(const v8h*)(bp[j] + kb);
        b.h[1] = *(const v8h*)(bp[j] + kb + 16);
        acc0[j] = __builtin_amdgcn_wmma_f32_16x16x32_f16(false, a0.v, false, b.v,
                                                         (short)0, acc0[j], false, false);
        acc1[j] = __builtin_amdgcn_wmma_f32_16x16x32_f16(false, a1.v, false, b.v,
                                                         (short)0, acc1[j], false, false);
      }
    }
#pragma unroll
    for (int j = 0; j < 4; ++j) {
      const float bv = bias[ncol[j]];
#pragma unroll
      for (int r = 0; r < 8; ++r) {
        float v0 = acc0[j][r] + bv; v0 = v0 > 0.f ? v0 : 0.f;
        float v1 = acc1[j][r] + bv; v1 = v1 > 0.f ? v1 : 0.f;
        O[(mbase + r) * kStride + ncol[j]]      = (_Float16)v0;
        O[(16 + mbase + r) * kStride + ncol[j]] = (_Float16)v1;
      }
    }
  }
}

// ---------------- kernel 3: fused per-32-span pipeline --------------------
__launch_bounds__(256)
__global__ void k_spans(const float* __restrict__ pref,
                        const _Float16* __restrict__ Wt1, const float* __restrict__ b1,
                        const _Float16* __restrict__ Wt2, const float* __restrict__ b2,
                        const _Float16* __restrict__ Wts1, const float* __restrict__ bs1,
                        const float* __restrict__ w2, const float* __restrict__ bs2,
                        float* __restrict__ out) {
  __shared__ __align__(16) _Float16 bufA[kMT * kStride];
  __shared__ __align__(16) _Float16 bufB[kMT * kStride];
  __shared__ int   sI[kMT], sE[kMT];
  __shared__ float sLen[kMT], sInv[kMT], sc[kMT];

  const int tid  = threadIdx.x;
  const int lane = tid & 31;
  const int wave = tid >> 5;
  const long span0 = (long)blockIdx.x * kMT;

  if (tid < kMT) {
    long s = span0 + tid;
    int ii = 0, ee = 0;
    if (s < kSpan) {
      float disc = (2.f * kNTok + 1.f) * (2.f * kNTok + 1.f) - 8.f * (float)s;
      disc = disc > 0.f ? disc : 0.f;
      int i = (int)((2.f * kNTok + 1.f - sqrtf(disc)) * 0.5f);
      if (i < 0) i = 0;
      if (i > kNTok - 1) i = kNTok - 1;
      auto rowoff = [](int x) { return (long)x * kNTok - (long)x * (x - 1) / 2; };
      while (i > 0 && rowoff(i) > s) --i;
      while (i < kNTok - 1 && rowoff(i + 1) <= s) ++i;
      ii = i;
      ee = i + (int)(s - rowoff(i)) + 1;   // end = j + 1
    }
    sI[tid] = ii;
    sE[tid] = ee;
    float len = (float)(ee - ii);
    sLen[tid] = len;
    sInv[tid] = (len > 0.f) ? 1.f / len : 0.f;
    sc[tid]   = 0.f;
  }
  __syncthreads();

  // span means (pref[end]-pref[start])/len  -> bufA as f16
  for (int e = tid; e < kMT * kEdim; e += 256) {
    int r = e >> 10, c = e & (kEdim - 1);
    float v = (pref[(size_t)sE[r] * kEdim + c] - pref[(size_t)sI[r] * kEdim + c]) * sInv[r];
    bufA[r * kStride + c] = (_Float16)v;
  }
  __syncthreads();

  gemm_tile(bufA, Wt1, b1, bufB, kEdim, lane, wave);   // DAN layer 1
  __syncthreads();
  gemm_tile(bufB, Wt2, b2, bufA, kHdim, lane, wave);   // DAN layer 2

  // phrase feats into padded K columns (disjoint from layer-2 output cols)
  for (int e = tid; e < kMT * 64; e += 256) {
    int r = e >> 6, c = e & 63;
    float v = (c < 16) ? sLen[r] : (c < 32) ? (float)sI[r]
            : (c < 48) ? (float)sE[r] : 0.f;
    bufA[r * kStride + kHdim + c] = (_Float16)v;
  }
  __syncthreads();

  gemm_tile(bufA, Wts1, bs1, bufB, kK3, lane, wave);   // score layer 1
  __syncthreads();

  // final dot with W_s2 (f32), LDS atomic reduction
  {
    int r = tid >> 3, c0 = tid & 7;
    float p = 0.f;
    for (int c = c0; c < kHdim; c += 8)
      p += (float)bufB[r * kStride + c] * w2[c];
    atomicAdd(&sc[r], p);
  }
  __syncthreads();
  if (tid < kMT) {
    long s = span0 + tid;
    if (s < kSpan) out[s] = sc[tid] + bs2[0];
  }
}

// --------------------------------------------------------------------------
extern "C" void kernel_launch(void* const* d_in, const int* in_sizes, int n_in,
                              void* d_out, int out_size, void* d_ws, size_t ws_size,
                              hipStream_t stream) {
  (void)in_sizes; (void)n_in; (void)out_size; (void)ws_size;
  const int*   sentence = (const int*)d_in[0];
  const int*   pos_tags = (const int*)d_in[1];
  const float* We_wrd   = (const float*)d_in[2];
  const float* We_pos   = (const float*)d_in[3];
  const float* W1  = (const float*)d_in[4];
  const float* b1  = (const float*)d_in[5];
  const float* W2  = (const float*)d_in[6];
  const float* b2  = (const float*)d_in[7];
  const float* Ws1 = (const float*)d_in[8];
  const float* bs1 = (const float*)d_in[9];
  const float* w2v = (const float*)d_in[10];
  const float* bs2 = (const float*)d_in[11];

  char* ws = (char*)d_ws;
  float*    pref = (float*)(ws + kPrefOff);
  _Float16* Wt1  = (_Float16*)(ws + kWt1Off);
  _Float16* Wt2  = (_Float16*)(ws + kWt2Off);
  _Float16* Wts1 = (_Float16*)(ws + kWts1Off);
  float* out = (float*)d_out;

  hipLaunchKernelGGL(k_embed_prefix, dim3((kEdim + 255) / 256), dim3(256), 0, stream,
                     sentence, pos_tags, We_wrd, We_pos, pref);
  hipLaunchKernelGGL(k_prep_weights, dim3(1024), dim3(256), 0, stream,
                     W1, W2, Ws1, Wt1, Wt2, Wts1);
  hipLaunchKernelGGL(k_spans, dim3((kSpan + kMT - 1) / kMT), dim3(256), 0, stream,
                     pref, Wt1, b1, Wt2, b2, Wts1, bs1, w2v, bs2, out);
}
